// SelfAttention_32650341384259
// MI455X (gfx1250) — compile-verified, matrix operand order checked by reading, not run on hardware
//
#include <hip/hip_runtime.h>
#include <hip/hip_bf16.h>

// ---------------------------------------------------------------------------
// MI455X (gfx1250) self-attention: bf16 WMMA pipeline (wave32) with
// Tensor-Data-Mover (TDM) double-buffered LDS staging.
//   x[4,2048,1024] --(bf16)--> QKV GEMMs --> flash attention --> out proj (f32)
// Matmuls: v_wmma_f32_16x16x32_bf16. Tile staging: tensor_load_to_lds
// (TENSORcnt + s_wait_tensorcnt), one barrier per K-step.
// ---------------------------------------------------------------------------

typedef __attribute__((ext_vector_type(16))) __bf16 v16bf;
typedef __attribute__((ext_vector_type(8)))  __bf16 v8bf;
typedef __attribute__((ext_vector_type(8)))  float  v8f;
typedef __attribute__((ext_vector_type(4)))  unsigned int u32x4;
typedef __attribute__((ext_vector_type(8)))  int i32x8;
typedef __attribute__((ext_vector_type(4)))  int i32x4;

#define E_DIM 1024
#define M_ROWS 8192   // B*T
#define T_SEQ 2048
#define H_HEADS 16
#define D_HEAD 64

__device__ __forceinline__ v8f vzero8() {
  v8f z;
#pragma unroll
  for (int i = 0; i < 8; ++i) z[i] = 0.0f;
  return z;
}

// CDNA5 16-bit A/B fragment: per lane, two contiguous 8-element (16B) chunks.
// p0 -> K[kbase .. kbase+7], p1 -> K[kbase+16 .. kbase+23], kbase = (lane>=16)*8.
__device__ __forceinline__ v16bf ld_frag16(const __bf16* p0, const __bf16* p1) {
  v8bf lo = *(const v8bf*)p0;
  v8bf hi = *(const v8bf*)p1;
  v16bf r;
#pragma unroll
  for (int i = 0; i < 8; ++i) { r[i] = lo[i]; r[i + 8] = hi[i]; }
  return r;
}

__device__ __forceinline__ v8f wmma_bf16(v16bf a, v16bf b, v8f c) {
  return __builtin_amdgcn_wmma_f32_16x16x32_bf16(false, a, false, b, (short)0, c,
                                                 false, false);
}

// ---------------------------------------------------------------------------
// TDM: DMA a 2D bf16 tile (tile_d0 elems/row x tile_d1 rows, row stride
// stride0 elems) from global into LDS, with hardware row padding so the LDS
// pitch becomes (row bytes + pad DWORDs*4).
//   pad_icode: row length in DWORDs = 2 << pad_icode
//   pad_acode: pad amount in DWORDs = pad_acode + 1
// Tracked by TENSORcnt; EXEC-independent (issue from one wave only).
// 6-arg builtin on this toolchain: (g0 v4u, g1 v8i, g2 v4i, g3 v4i, v8i, cpol).
// ---------------------------------------------------------------------------
__device__ __forceinline__ void tdm_load_2d_bf16(const void* gptr, unsigned lds_addr,
                                                 unsigned tile_d0, unsigned tile_d1,
                                                 unsigned long long stride0,
                                                 unsigned pad_icode, unsigned pad_acode) {
  unsigned long long ga = (unsigned long long)(uintptr_t)gptr;
  u32x4 g0;
  g0[0] = 1u;                                   // count=1, user descriptor
  g0[1] = lds_addr;                             // LDS byte address of tile start
  g0[2] = (unsigned)ga;                         // global_addr[31:0]
  g0[3] = (unsigned)(ga >> 32) | (2u << 30);    // global_addr[56:32] | type=2
  const unsigned dim0 = 0x7FFFFFFFu, dim1 = 0x7FFFFFFFu;  // OOB-only: never clip
  i32x8 g1;
  g1[0] = (int)((1u << 16) |                    // data_size = 2 bytes
                (1u << 20) |                    // pad_enable
                (pad_icode << 22) | (pad_acode << 25));
  g1[1] = (int)((dim0 & 0xFFFFu) << 16);                     // tensor_dim0 lo
  g1[2] = (int)((dim0 >> 16) | ((dim1 & 0xFFFFu) << 16));    // dim0 hi | dim1 lo
  g1[3] = (int)((dim1 >> 16) | (tile_d0 << 16));             // dim1 hi | tile_dim0
  g1[4] = (int)(tile_d1);                                    // tile_dim1 | tile_dim2=0
  g1[5] = (int)(unsigned)(stride0 & 0xFFFFFFFFu);            // dim0_stride lo
  g1[6] = (int)(unsigned)((stride0 >> 32) & 0xFFFFu);        // stride hi | stride1 lo
  g1[7] = 0;
  i32x4 gz;
  gz[0] = gz[1] = gz[2] = gz[3] = 0;            // groups 2/3: unused (2D tile)
  i32x8 gz8;
#pragma unroll
  for (int i = 0; i < 8; ++i) gz8[i] = 0;       // unused trailing group (VADDR4)
  __builtin_amdgcn_tensor_load_to_lds(g0, g1, gz, gz, gz8, 0);
}

__device__ __forceinline__ unsigned lds_addr_of(const void* p) {
  return (unsigned)(uintptr_t)p;  // low 32 bits of flat LDS addr = segment offset
}

// ---------------------------------------------------------------------------
// Elementwise f32 -> bf16 convert
// ---------------------------------------------------------------------------
__global__ __launch_bounds__(256) void cvt_bf16_kernel(const float* __restrict__ in,
                                                       __bf16* __restrict__ out,
                                                       int n) {
  int i = blockIdx.x * 256 + threadIdx.x;
  if (i < n) out[i] = (__bf16)in[i];
}

// ---------------------------------------------------------------------------
// 1024x1024 transpose + convert: out[n*1024+k] = (bf16) in[k*1024+n]
// ---------------------------------------------------------------------------
__global__ __launch_bounds__(256) void transpose_cvt_kernel(const float* __restrict__ in,
                                                            __bf16* __restrict__ out) {
  __shared__ float tile[32][33];
  const int bx = blockIdx.x * 32;
  const int by = blockIdx.y * 32;
  const int tx = threadIdx.x & 31;
  const int ty = threadIdx.x >> 5;
#pragma unroll
  for (int i = 0; i < 4; ++i)
    tile[ty + i * 8][tx] = in[(size_t)(by + ty + i * 8) * E_DIM + bx + tx];
  __syncthreads();
#pragma unroll
  for (int i = 0; i < 4; ++i)
    out[(size_t)(bx + ty + i * 8) * E_DIM + by + tx] = (__bf16)tile[tx][ty + i * 8];
}

// ---------------------------------------------------------------------------
// GEMM: C[M=8192, N=1024] = A(bf16)[M,K] x BT(bf16)[N,K]^T + bias, * oscale
// Block 256 thr = 8 waves (2x4), wave tile 64x64, WG tile 128x256, K-step 32.
// TDM double-buffered staging: wave0 issues DMA for tile i+1 right after the
// barrier that publishes tile i; everyone computes tile i meanwhile.
// MODE 0: f32 out [M,1024]; MODE 1: bf16 -> [B,H,T,D]; MODE 2: bf16 -> [B,H,D,T]
// ---------------------------------------------------------------------------
template <int MODE>
__global__ __launch_bounds__(256) void gemm_bf16_kernel(const __bf16* __restrict__ A,
                                                        const __bf16* __restrict__ BT,
                                                        const float* __restrict__ bias,
                                                        void* __restrict__ outp,
                                                        float oscale) {
  constexpr int LDA = 40;  // 64B row + 16B TDM pad
  constexpr int LDB = 40;
  constexpr int NIT = E_DIM / 32;
  __shared__ __align__(16) __bf16 ldsA[2][128 * LDA];
  __shared__ __align__(16) __bf16 ldsB[2][256 * LDB];

  const int tid = threadIdx.x;
  const int lane = tid & 31;
  const int wave = tid >> 5;
  const int waveM = wave >> 2;
  const int waveN = wave & 3;
  const int mBlk = blockIdx.x * 128;
  const int nBlk = blockIdx.y * 256;
  const int lm = lane & 15;
  const int kb = (lane >> 4) * 8;

  v8f acc[4][4];
#pragma unroll
  for (int i = 0; i < 4; ++i)
#pragma unroll
    for (int j = 0; j < 4; ++j) acc[i][j] = vzero8();

  // Prologue: DMA tile 0 into buffer 0 (rows x 32 cols, pad 16DW->+4DW).
  if (wave == 0) {
    tdm_load_2d_bf16(A + (size_t)mBlk * E_DIM, lds_addr_of(&ldsA[0][0]),
                     32, 128, E_DIM, 3, 3);
    tdm_load_2d_bf16(BT + (size_t)nBlk * E_DIM, lds_addr_of(&ldsB[0][0]),
                     32, 256, E_DIM, 3, 3);
  }

  for (int it = 0; it < NIT; ++it) {
    const int p = it & 1;
    if (wave == 0) __builtin_amdgcn_s_wait_tensorcnt(0);  // tile it landed
    __syncthreads();                                      // publish to all waves
    if (wave == 0 && it + 1 < NIT) {                      // overlap next DMA
      const int k0n = (it + 1) * 32;
      tdm_load_2d_bf16(A + (size_t)mBlk * E_DIM + k0n, lds_addr_of(&ldsA[1 - p][0]),
                       32, 128, E_DIM, 3, 3);
      tdm_load_2d_bf16(BT + (size_t)nBlk * E_DIM + k0n, lds_addr_of(&ldsB[1 - p][0]),
                       32, 256, E_DIM, 3, 3);
    }

    v16bf af[4], bfr[4];
#pragma unroll
    for (int mi = 0; mi < 4; ++mi) {
      const __bf16* pp = &ldsA[p][(waveM * 64 + mi * 16 + lm) * LDA + kb];
      af[mi] = ld_frag16(pp, pp + 16);
    }
#pragma unroll
    for (int ni = 0; ni < 4; ++ni) {
      const __bf16* pp = &ldsB[p][(waveN * 64 + ni * 16 + lm) * LDB + kb];
      bfr[ni] = ld_frag16(pp, pp + 16);
    }
#pragma unroll
    for (int mi = 0; mi < 4; ++mi)
#pragma unroll
      for (int ni = 0; ni < 4; ++ni)
        acc[mi][ni] = wmma_bf16(af[mi], bfr[ni], acc[mi][ni]);
  }

  // Store. C layout: vgpr r, lane l -> M = r + 8*(l>>4), N = l&15 within tile.
  float* outF = (float*)outp;
  __bf16* outB = (__bf16*)outp;
  const int hf8 = (lane >> 4) * 8;
#pragma unroll
  for (int mi = 0; mi < 4; ++mi)
#pragma unroll
    for (int ni = 0; ni < 4; ++ni) {
      const int N = nBlk + waveN * 64 + ni * 16 + lm;
      const float bcol = bias[N];
      const int Mbase = mBlk + waveM * 64 + mi * 16 + hf8;
#pragma unroll
      for (int r = 0; r < 8; ++r) {
        const int M = Mbase + r;
        const float val = (acc[mi][ni][r] + bcol) * oscale;
        if (MODE == 0) {
          outF[(size_t)M * E_DIM + N] = val;
        } else {
          const int b = M >> 11, t = M & (T_SEQ - 1);
          const int h = N >> 6, d = N & (D_HEAD - 1);
          if (MODE == 1)
            outB[(((size_t)(b * H_HEADS + h)) * T_SEQ + t) * D_HEAD + d] = (__bf16)val;
          else
            outB[(((size_t)(b * H_HEADS + h)) * D_HEAD + d) * T_SEQ + t] = (__bf16)val;
        }
      }
    }
}

// ---------------------------------------------------------------------------
// Flash attention. Grid: (T/128, B*H). Block 256 thr = 8 waves; wave w owns
// 16 query rows. TDM double-buffers the K / V^T tiles. Q is pre-scaled by
// 1/sqrt(D) at projection time, so logits = q.k directly.
// q,k: [B,H,T,D] bf16; vt: [B,H,D,T] bf16; y: [B,T,E] bf16.
// ---------------------------------------------------------------------------
__global__ __launch_bounds__(256) void attn_kernel(const __bf16* __restrict__ q,
                                                   const __bf16* __restrict__ k,
                                                   const __bf16* __restrict__ vt,
                                                   __bf16* __restrict__ y) {
  constexpr int LDK = 72;   // 128B row + 16B TDM pad
  constexpr int LDV = 136;  // 256B row + 16B TDM pad
  constexpr int LDP = 136;
  constexpr int NIT = T_SEQ / 128;
  __shared__ __align__(16) __bf16 ldsK[2][128 * LDK];
  __shared__ __align__(16) __bf16 ldsV[2][64 * LDV];
  __shared__ __align__(16) __bf16 ldsP[128 * LDP];

  const int tid = threadIdx.x;
  const int lane = tid & 31;
  const int w = tid >> 5;
  const int bh = blockIdx.y;
  const int b = bh >> 4, h = bh & 15;
  const int qBase = blockIdx.x * 128;
  const int lm = lane & 15;
  const int hf = lane >> 4;
  const int kb = hf * 8;

  // Q fragments (registers for the whole kernel).
  const __bf16* qrow = q + ((size_t)bh * T_SEQ + qBase + w * 16 + lm) * D_HEAD;
  v16bf aq[2];
#pragma unroll
  for (int ks = 0; ks < 2; ++ks)
    aq[ks] = ld_frag16(qrow + ks * 32 + kb, qrow + ks * 32 + kb + 16);

  v8f o[4];
#pragma unroll
  for (int i = 0; i < 4; ++i) o[i] = vzero8();
  float mrow[8], lrow[8];
#pragma unroll
  for (int r = 0; r < 8; ++r) { mrow[r] = -3.0e38f; lrow[r] = 0.0f; }

  const __bf16* kBase = k + (size_t)bh * T_SEQ * D_HEAD;
  const __bf16* vBase = vt + (size_t)bh * D_HEAD * T_SEQ;

  // Prologue: DMA K/V^T tiles for t0 = 0 into buffer 0.
  if (w == 0) {
    tdm_load_2d_bf16(kBase, lds_addr_of(&ldsK[0][0]), 64, 128, D_HEAD, 4, 3);
    tdm_load_2d_bf16(vBase, lds_addr_of(&ldsV[0][0]), 128, 64, T_SEQ, 5, 3);
  }

  for (int it = 0; it < NIT; ++it) {
    const int p = it & 1;
    if (w == 0) __builtin_amdgcn_s_wait_tensorcnt(0);
    __syncthreads();
    if (w == 0 && it + 1 < NIT) {
      const int t0n = (it + 1) * 128;
      tdm_load_2d_bf16(kBase + (size_t)t0n * D_HEAD, lds_addr_of(&ldsK[1 - p][0]),
                       64, 128, D_HEAD, 4, 3);
      tdm_load_2d_bf16(vBase + t0n, lds_addr_of(&ldsV[1 - p][0]),
                       128, 64, T_SEQ, 5, 3);
    }

    // S = Q K^T  (8 column tiles x 2 K-steps)
    v8f s[8];
#pragma unroll
    for (int ni = 0; ni < 8; ++ni) {
      const __bf16* pp = &ldsK[p][(ni * 16 + lm) * LDK + kb];
      v16bf b0 = ld_frag16(pp, pp + 16);
      v16bf b1 = ld_frag16(pp + 32, pp + 48);
      v8f z = vzero8();
      z = wmma_bf16(aq[0], b0, z);
      s[ni] = wmma_bf16(aq[1], b1, z);
    }

    // Online softmax per row (row lives in one 16-lane half; reduce via shfl).
#pragma unroll
    for (int r = 0; r < 8; ++r) {
      float mx = -3.0e38f;
#pragma unroll
      for (int ni = 0; ni < 8; ++ni) mx = fmaxf(mx, s[ni][r]);
#pragma unroll
      for (int d2 = 8; d2 >= 1; d2 >>= 1) mx = fmaxf(mx, __shfl_xor(mx, d2, 32));
      const float mnew = fmaxf(mrow[r], mx);
      const float corr = __expf(mrow[r] - mnew);
      float sum = 0.0f;
#pragma unroll
      for (int ni = 0; ni < 8; ++ni) {
        float pv = __expf(s[ni][r] - mnew);
        s[ni][r] = pv;
        sum += pv;
      }
#pragma unroll
      for (int d2 = 8; d2 >= 1; d2 >>= 1) sum += __shfl_xor(sum, d2, 32);
      lrow[r] = lrow[r] * corr + sum;
      mrow[r] = mnew;
#pragma unroll
      for (int di = 0; di < 4; ++di) o[di][r] *= corr;
    }

    // P (C layout) -> LDS bf16, re-read as A fragments.
#pragma unroll
    for (int ni = 0; ni < 8; ++ni)
#pragma unroll
      for (int r = 0; r < 8; ++r)
        ldsP[(w * 16 + r + 8 * hf) * LDP + ni * 16 + lm] = (__bf16)s[ni][r];
    __syncthreads();

    // O += P V  (4 d-tiles x 4 K-steps over 128 keys)
    v16bf ap[4];
#pragma unroll
    for (int ks = 0; ks < 4; ++ks) {
      const __bf16* pp = &ldsP[(w * 16 + lm) * LDP + ks * 32 + kb];
      ap[ks] = ld_frag16(pp, pp + 16);
    }
#pragma unroll
    for (int di = 0; di < 4; ++di)
#pragma unroll
      for (int ks = 0; ks < 4; ++ks) {
        const __bf16* pp = &ldsV[p][(di * 16 + lm) * LDV + ks * 32 + kb];
        o[di] = wmma_bf16(ap[ks], ld_frag16(pp, pp + 16), o[di]);
      }
  }

  // Normalize and store y[b, t, h*64 + d] as bf16.
#pragma unroll
  for (int r = 0; r < 8; ++r) {
    const float inv = 1.0f / lrow[r];
    const int qr = qBase + w * 16 + r + 8 * hf;
#pragma unroll
    for (int di = 0; di < 4; ++di)
      y[((size_t)b * T_SEQ + qr) * E_DIM + h * D_HEAD + di * 16 + lm] =
          (__bf16)(o[di][r] * inv);
  }
}

// ---------------------------------------------------------------------------
// Host launcher
// ---------------------------------------------------------------------------
extern "C" void kernel_launch(void* const* d_in, const int* in_sizes, int n_in,
                              void* d_out, int out_size, void* d_ws, size_t ws_size,
                              hipStream_t stream) {
  (void)in_sizes; (void)n_in; (void)out_size; (void)ws_size;
  const float* x  = (const float*)d_in[0];
  const float* Wq = (const float*)d_in[1];
  const float* bq = (const float*)d_in[2];
  const float* Wk = (const float*)d_in[3];
  const float* bk = (const float*)d_in[4];
  const float* Wv = (const float*)d_in[5];
  const float* bv = (const float*)d_in[6];
  const float* Wp = (const float*)d_in[7];
  const float* bp = (const float*)d_in[8];
  float* out = (float*)d_out;

  char* ws = (char*)d_ws;
  const size_t MB = 1024 * 1024;
  // Layout (72 MB): Xbf (reused as y after attention) | 4x W^T | Q | K | V^T
  __bf16* Xbf = (__bf16*)(ws);             // 16 MB: [8192,1024] bf16 (also y)
  __bf16* WqT = (__bf16*)(ws + 16 * MB);   // 2 MB each
  __bf16* WkT = (__bf16*)(ws + 18 * MB);
  __bf16* WvT = (__bf16*)(ws + 20 * MB);
  __bf16* WpT = (__bf16*)(ws + 22 * MB);
  __bf16* Qb  = (__bf16*)(ws + 24 * MB);   // 16 MB [B,H,T,D] (pre-scaled by 1/8)
  __bf16* Kb  = (__bf16*)(ws + 40 * MB);   // 16 MB [B,H,T,D]
  __bf16* Vt  = (__bf16*)(ws + 56 * MB);   // 16 MB [B,H,D,T]

  const int nx = M_ROWS * E_DIM;
  cvt_bf16_kernel<<<nx / 256, 256, 0, stream>>>(x, Xbf, nx);
  dim3 tg(32, 32);
  transpose_cvt_kernel<<<tg, 256, 0, stream>>>(Wq, WqT);
  transpose_cvt_kernel<<<tg, 256, 0, stream>>>(Wk, WkT);
  transpose_cvt_kernel<<<tg, 256, 0, stream>>>(Wv, WvT);
  transpose_cvt_kernel<<<tg, 256, 0, stream>>>(Wp, WpT);

  dim3 gg(M_ROWS / 128, E_DIM / 256);  // (64, 4)
  gemm_bf16_kernel<1><<<gg, 256, 0, stream>>>(Xbf, WqT, bq, (void*)Qb, 0.125f);
  gemm_bf16_kernel<1><<<gg, 256, 0, stream>>>(Xbf, WkT, bk, (void*)Kb, 1.0f);
  gemm_bf16_kernel<2><<<gg, 256, 0, stream>>>(Xbf, WvT, bv, (void*)Vt, 1.0f);

  attn_kernel<<<dim3(T_SEQ / 128, 4 * H_HEADS), 256, 0, stream>>>(Qb, Kb, Vt, Xbf);

  gemm_bf16_kernel<0><<<gg, 256, 0, stream>>>(Xbf, WpT, bp, (void*)out, 1.0f);
}
